// LinkPredictor_34230889349178
// MI455X (gfx1250) — matrix operand order repaired and checked by reading, not compiled, not run on hardware
//
#include <hip/hip_runtime.h>
#include <hip/hip_bf16.h>

typedef _Float16 v16h __attribute__((ext_vector_type(16)));
typedef float    v8f  __attribute__((ext_vector_type(8)));

#define HID   256
#define KDIM  512           // 2*HID
#define TPB   512           // 16 waves of 32
#define W1_LDS_BYTES (KDIM * HID * 2)          // 262144: f16 W1 fragments
#define A_LDS_BYTES  (16 * 32 * 32)            // 16384: 16 K-tiles x 32 lanes x 32B
#define RED_LDS_BYTES (16 * 16 * 4)            // 1024
#define SMEM_BYTES (W1_LDS_BYTES + A_LDS_BYTES + RED_LDS_BYTES)

// ---------------------------------------------------------------------------
// Pre-pack W1 [256 x 512] f32 (row n = out-channel, col k = in-channel) into
// f16 WMMA B-fragments: frag(kt,nt) lane L holds B[k][n] = W1[n][k] with
//   n = nt*16 + (L%16),  k = kt*32 + (L/16)*16 + h   (h = 0..15, contiguous!)
// Linear order: frag = kt*512 + nt*32 + L, 16 halves (32B) per frag-lane.
// ---------------------------------------------------------------------------
__global__ void pack_w1_kernel(const float* __restrict__ W1,
                               _Float16* __restrict__ w1p) {
    int t = blockIdx.x * blockDim.x + threadIdx.x;   // 0..8191
    if (t >= 16 * 16 * 32) return;
    int lane = t & 31;
    int nt   = (t >> 5) & 15;
    int kt   = t >> 9;
    int n = nt * 16 + (lane & 15);
    int k = kt * 32 + (lane >> 4) * 16;
    const float* src = W1 + (long)n * KDIM + k;
    _Float16* dst = w1p + (long)t * 16;
#pragma unroll
    for (int i = 0; i < 16; ++i) dst[i] = (_Float16)src[i];
}

// ---------------------------------------------------------------------------
// Main fused kernel: gather+concat -> f16 -> WMMA GEMM -> bias+ReLU -> dot W2
// ---------------------------------------------------------------------------
__global__ __launch_bounds__(TPB)
void linkpred_kernel(const float* __restrict__ xg,   // [50000 x 256]
                     const float* __restrict__ xd,   // [20000 x 256]
                     const int*   __restrict__ eidx, // [2 x E]
                     const _Float16* __restrict__ w1p,
                     const float* __restrict__ b1,
                     const float* __restrict__ w2,
                     const float* __restrict__ b2,
                     float* __restrict__ out,
                     int E, int nBlocks) {
    extern __shared__ char smem[];
    _Float16* sW1 = (_Float16*)smem;                           // 256 KB
    char*     sA  = smem + W1_LDS_BYTES;                       // 16 KB
    float*    sRed = (float*)(smem + W1_LDS_BYTES + A_LDS_BYTES);

    const int tid  = threadIdx.x;
    const int lane = tid & 31;
    const int wv   = tid >> 5;                // 0..15 : this wave's N-tile

    // ---- stage packed W1 into LDS (once per workgroup) ----
    {
        const float4* src = (const float4*)w1p;
        float4*       dst = (float4*)sW1;
        for (int i = tid; i < W1_LDS_BYTES / 16; i += TPB) dst[i] = src[i];
    }
    __syncthreads();

    const float b2v = b2[0];
    const float b1v = b1[wv * 16 + (lane & 15)];
    const float w2v = w2[wv * 16 + (lane & 15)];

    for (int blk = blockIdx.x; blk < nBlocks; blk += gridDim.x) {
        // ---- gather 16 edges, convert f32->f16, scatter into A-fragment layout
        // chunk c: row m = c/64, 8 consecutive K at j = c%64 (K = j*8..j*8+7)
        for (int c = tid; c < 1024; c += TPB) {
            int m = c >> 6;
            int j = c & 63;
            int e = blk * 16 + m;
            struct alignas(16) H8 { _Float16 h[8]; } pk;
            if (e < E) {
                const float* src;
                if (j < 32) {                       // gene half (K 0..255)
                    int r = eidx[e];
                    src = xg + (long)r * HID + j * 8;
                } else {                            // disease half (K 256..511)
                    int d = eidx[E + e];
                    src = xd + (long)d * HID + (j - 32) * 8;
                }
                float f[8];
                *(float4*)&f[0] = ((const float4*)src)[0];
                *(float4*)&f[4] = ((const float4*)src)[1];
#pragma unroll
                for (int i = 0; i < 8; ++i) pk.h[i] = (_Float16)f[i];
            } else {
#pragma unroll
                for (int i = 0; i < 8; ++i) pk.h[i] = (_Float16)0.0f;
            }
            // contiguous A 16x32 layout: rel 0 -> lane m bytes 0-15,
            // rel 8 -> lane m bytes 16-31, rel 16 -> lane m+16 bytes 0-15,
            // rel 24 -> lane m+16 bytes 16-31
            int kt  = j >> 2;
            int rel = (j & 3) * 8;
            int dl  = m + ((rel & 16) ? 16 : 0);
            int off = (kt * 32 + dl) * 32 + ((rel & 8) ? 16 : 0);
            *(H8*)(sA + off) = pk;
        }
        __syncthreads();

        // ---- 16 x WMMA f32_16x16x32_f16 over K=512 (A,B from LDS) ----
        v8f acc = {};
        const v16h* Af = (const v16h*)sA;
        const v16h* Bf = (const v16h*)sW1;
#pragma unroll
        for (int kt = 0; kt < 16; ++kt) {
            v16h a = Af[kt * 32 + lane];
            v16h b = Bf[(kt * 16 + wv) * 32 + lane];
            acc = __builtin_amdgcn_wmma_f32_16x16x32_f16(
                false, a, false, b, (short)0, acc, false, false);
        }

        // ---- epilogue: bias + ReLU + dot with W2 slice, reduce over lanes ----
        // acc[v] = H[m][n], n = wv*16 + lane%16, m = v + 8*(lane/16)
#pragma unroll
        for (int v = 0; v < 8; ++v) {
            float p = fmaxf(acc[v] + b1v, 0.0f) * w2v;
            p += __shfl_xor(p, 1, 32);
            p += __shfl_xor(p, 2, 32);
            p += __shfl_xor(p, 4, 32);
            p += __shfl_xor(p, 8, 32);      // stays inside 16-lane halves
            if ((lane & 15) == 0) {
                int m = v + ((lane >= 16) ? 8 : 0);
                sRed[wv * 16 + m] = p;
            }
        }
        __syncthreads();

        // ---- cross-wave sum + b2, write out ----
        if (tid < 16) {
            float s = b2v;
#pragma unroll
            for (int w = 0; w < 16; ++w) s += sRed[w * 16 + tid];
            int e = blk * 16 + tid;
            if (e < E) out[e] = s;
        }
        __syncthreads();   // protect sA / sRed for next block iteration
    }
}

extern "C" void kernel_launch(void* const* d_in, const int* in_sizes, int n_in,
                              void* d_out, int out_size, void* d_ws, size_t ws_size,
                              hipStream_t stream) {
    const float* xg   = (const float*)d_in[0];
    const float* xd   = (const float*)d_in[1];
    const int*   eidx = (const int*)  d_in[2];   // [2 x E] int32
    const float* W1   = (const float*)d_in[3];
    const float* b1   = (const float*)d_in[4];
    const float* w2   = (const float*)d_in[5];
    const float* b2   = (const float*)d_in[6];
    float* out = (float*)d_out;

    int E = in_sizes[2] / 2;
    int nBlocks = (E + 15) / 16;

    _Float16* w1p = (_Float16*)d_ws;             // 256 KB packed W1

    pack_w1_kernel<<<(16 * 16 * 32 + 255) / 256, 256, 0, stream>>>(W1, w1p);

    (void)hipFuncSetAttribute((const void*)linkpred_kernel,
                              hipFuncAttributeMaxDynamicSharedMemorySize,
                              (int)SMEM_BYTES);

    int grid = nBlocks < 512 ? nBlocks : 512;    // persistent workgroups
    linkpred_kernel<<<grid, TPB, SMEM_BYTES, stream>>>(
        xg, xd, eidx, w1p, b1, w2, b2, out, E, nBlocks);
}
